// CrossAttention_38637525795303
// MI455X (gfx1250) — matrix operand order repaired and checked by reading, not compiled, not run on hardware
//
#include <hip/hip_runtime.h>
#include <hip/hip_bf16.h>
#include <math.h>

// ---------------------------------------------------------------------------
// CDNA5 / gfx1250 implementation of the cross-attention + FFN block.
// Heavy math: v_wmma_f32_16x16x32_bf16 (bf16 in, f32 acc).
// Tile staging: GLOBAL_LOAD_ASYNC_TO_LDS_B128 (ASYNCcnt) when the toolchain
// exposes the builtin, else a register-staged software pipeline.
// ---------------------------------------------------------------------------

typedef __bf16 bf16;
typedef __attribute__((ext_vector_type(16))) bf16  v16bf;
typedef __attribute__((ext_vector_type(8)))  bf16  v8bf;
typedef __attribute__((ext_vector_type(8)))  float v8f;
typedef __attribute__((ext_vector_type(4)))  int   v4i;

#define AS1 __attribute__((address_space(1)))
#define AS3 __attribute__((address_space(3)))

#if defined(__AMDGCN__) &&                                                    \
    __has_builtin(__builtin_amdgcn_global_load_async_to_lds_b128) &&          \
    __has_builtin(__builtin_amdgcn_s_wait_asynccnt)
#define ASYNC_LDS 1
#else
#define ASYNC_LDS 0
#endif

#define BSQ   2048   // BS1 == BS2
#define EDIM  1024
#define FF    4096
#define NHEAD 8
#define HDIM  128

// ------------------------- f32 -> bf16 convert -----------------------------
__global__ void cvt_f32_bf16(const float* __restrict__ src,
                             bf16* __restrict__ dst, long n) {
  long i = (long)blockIdx.x * blockDim.x + threadIdx.x;
  if (i < n) dst[i] = (bf16)src[i];
}

// ------------------------- bf16 2-D transpose ------------------------------
// dst[c][m] = src[m][c]   (rows x cols -> cols x rows), tiles of 32x32
__global__ __launch_bounds__(256)
void transpose_bf16(const bf16* __restrict__ src, bf16* __restrict__ dst,
                    int rows, int cols) {
  __shared__ bf16 t[32][33];
  const int bx = blockIdx.x * 32;        // col base
  const int by = blockIdx.y * 32;        // row base
  const int tx = threadIdx.x & 31;
  const int ty = threadIdx.x >> 5;       // 0..7
#pragma unroll
  for (int i = 0; i < 4; ++i) {
    int rr = ty + i * 8;
    t[rr][tx] = src[(long)(by + rr) * cols + bx + tx];
  }
  __syncthreads();
#pragma unroll
  for (int i = 0; i < 4; ++i) {
    int rr = ty + i * 8;                 // local column index
    dst[(long)(bx + rr) * rows + by + tx] = t[tx][rr];
  }
}

// --------------------- softmax over the HEADS axis -------------------------
// scores layout: [h][n][m] planes of `plane` elements; softmax across the 8
// planes at each (n,m) point. In-place bf16. Purely pointwise (no m-reduce).
__global__ void softmax_heads(bf16* __restrict__ s, long plane) {
  long i = (long)blockIdx.x * blockDim.x + threadIdx.x;
  if (i >= plane) return;
  float v[NHEAD];
  float mx = -1e30f;
#pragma unroll
  for (int h = 0; h < NHEAD; ++h) {
    v[h] = (float)s[(long)h * plane + i];
    mx = fmaxf(mx, v[h]);
  }
  float sum = 0.f;
#pragma unroll
  for (int h = 0; h < NHEAD; ++h) { v[h] = __expf(v[h] - mx); sum += v[h]; }
  const float inv = 1.f / sum;
#pragma unroll
  for (int h = 0; h < NHEAD; ++h)
    s[(long)h * plane + i] = (bf16)(v[h] * inv);
}

// --------------------- residual add + LayerNorm ----------------------------
// out = LN(x + y) * g + b ; one row (EDIM=1024) per block, 256 threads.
template <bool OUTBF16>
__global__ __launch_bounds__(256)
void add_layernorm(const float* __restrict__ x, const float* __restrict__ y,
                   const float* __restrict__ g, const float* __restrict__ b,
                   float* __restrict__ outf, bf16* __restrict__ outb) {
  __shared__ float red[256];
  const long base = (long)blockIdx.x * EDIM;
  const int tid = threadIdx.x;
  float vals[4], s = 0.f, ss = 0.f;
#pragma unroll
  for (int i = 0; i < 4; ++i) {
    int c = tid + i * 256;
    float v = x[base + c] + y[base + c];
    vals[i] = v; s += v; ss += v * v;
  }
  red[tid] = s; __syncthreads();
  for (int o = 128; o > 0; o >>= 1) {
    if (tid < o) red[tid] += red[tid + o];
    __syncthreads();
  }
  const float mean = red[0] * (1.f / EDIM);
  __syncthreads();
  red[tid] = ss; __syncthreads();
  for (int o = 128; o > 0; o >>= 1) {
    if (tid < o) red[tid] += red[tid + o];
    __syncthreads();
  }
  const float var = red[0] * (1.f / EDIM) - mean * mean;
  const float inv = rsqrtf(var + 1e-5f);
#pragma unroll
  for (int i = 0; i < 4; ++i) {
    int c = tid + i * 256;
    float o = (vals[i] - mean) * inv * g[c] + b[c];
    outf[base + c] = o;
    if (OUTBF16) outb[base + c] = (bf16)o;
  }
}

// --------------------------- WMMA GEMM -------------------------------------
// C[M,N] = alpha * (A[M,K] @ W[N,K]^T) (+ bias[N]) (ReLU) ; bf16 in, f32 acc.
// Batched over blockIdx.z with per-z element strides (per-head slices).
// Block tile 128x64, BK=32; 8 waves, each wave owns a 32x32 tile (2x2 WMMA).
//
// Fragment layouts per CDNA5 ISA 7.12.2 (16-bit, wave32):
//   A 16x32 : lane<16 row=lane  K:{0..7,16..23} | lane>=16 row=lane-16 K:{8..15,24..31}
//   B 32x16 : lane n holds 16 K in A's order (lane<16 -> K{0..7,16..23}, else K{8..15,24..31})
//             == one row of the row-major weight W[N,K]  -> no transpose needed.
__device__ __forceinline__ v16bf cat8(v8bf lo, v8bf hi) {
  v16bf r;
#pragma unroll
  for (int i = 0; i < 8; ++i) { r[i] = lo[i]; r[i + 8] = hi[i]; }
  return r;
}

template <bool BIAS, bool RELU, bool OUTF32, bool OUTBF16>
__global__ __launch_bounds__(256)
void gemm_wmma(const bf16* __restrict__ A, const bf16* __restrict__ B,
               const float* __restrict__ bias,
               float* __restrict__ Cf, bf16* __restrict__ Cb,
               int K, int lda, int ldb, int ldc,
               long sAz, long sBz, long sCz, float alpha) {
  // padded rows (40 elem = 80 B) keep every fragment ds-load 16B aligned and
  // spread across banks
#if ASYNC_LDS
  __shared__ bf16 sA[2][128 * 40];
  __shared__ bf16 sB[2][64 * 40];
#else
  __shared__ bf16 sA[1][128 * 40];
  __shared__ bf16 sB[1][64 * 40];
#endif

  const int tid  = threadIdx.x;
  const int lane = tid & 31;
  const int wave = tid >> 5;
  const int wm   = wave & 3;            // 4 row groups of 32
  const int wn   = wave >> 2;           // 2 col groups of 32
  const long z   = blockIdx.z;
  const bf16* Az = A + z * sAz;
  const bf16* Bz = B + z * sBz;
  const long  cz = z * sCz;
  const int   bm = blockIdx.y * 128;
  const int   bn = blockIdx.x * 64;

  // per-thread staging slots: A tile 128x32 = 2 chunks of 8 bf16, B tile 64x32 = 1
  const int rA0 = tid >> 2;             // 0..63
  const int rA1 = rA0 + 64;             // 64..127
  const int cc  = (tid & 3) << 3;       // 0/8/16/24
  const bf16* gA0 = Az + (long)(bm + rA0) * lda + cc;
  const bf16* gA1 = Az + (long)(bm + rA1) * lda + cc;
  const bf16* gB  = Bz + (long)(bn + rA0) * ldb + cc;

  v8f acc[2][2] = {};
  const int ar = lane & 15;             // A row / B col / epilogue N index
  const int ak = (lane >> 4) << 3;      // A K sub-offset: 0 or 8
  const int bk = (lane >> 4) << 3;      // B K sub-offset: 0 or 8

#if ASYNC_LDS
  // ---- async DMA straight to LDS (no VGPR round-trip), double buffered ----
  auto issue = [&](int bufi, int k0) {
    __builtin_amdgcn_global_load_async_to_lds_b128(
        (AS1 v4i*)(gA0 + k0), (AS3 v4i*)&sA[bufi][rA0 * 40 + cc], 0, 0);
    __builtin_amdgcn_global_load_async_to_lds_b128(
        (AS1 v4i*)(gA1 + k0), (AS3 v4i*)&sA[bufi][rA1 * 40 + cc], 0, 0);
    __builtin_amdgcn_global_load_async_to_lds_b128(
        (AS1 v4i*)(gB + k0), (AS3 v4i*)&sB[bufi][rA0 * 40 + cc], 0, 0);
  };
  issue(0, 0);
  int buf = 0;
#else
  // ---- register-staged software pipeline ----
  v8bf ra0 = *(const v8bf*)gA0;
  v8bf ra1 = *(const v8bf*)gA1;
  v8bf rb  = *(const v8bf*)gB;
#endif

  for (int k0 = 0; k0 < K; k0 += 32) {
    const bool has_next = (k0 + 32) < K;
#if ASYNC_LDS
    if (has_next) {
      issue(buf ^ 1, k0 + 32);                   // prefetch next tile (other buf)
      __builtin_amdgcn_s_wait_asynccnt(3);       // current tile's 3 copies landed
    } else {
      __builtin_amdgcn_s_wait_asynccnt(0);
    }
    __syncthreads();
    const bf16* pa = sA[buf];
    const bf16* pb = sB[buf];
#else
    *(v8bf*)&sA[0][rA0 * 40 + cc] = ra0;
    *(v8bf*)&sA[0][rA1 * 40 + cc] = ra1;
    *(v8bf*)&sB[0][rA0 * 40 + cc] = rb;
    __syncthreads();
    if (has_next) {                              // issue next tile loads early;
      ra0 = *(const v8bf*)(gA0 + k0 + 32);       // loadcnt wait is deferred to
      ra1 = *(const v8bf*)(gA1 + k0 + 32);       // the next iteration's stores
      rb  = *(const v8bf*)(gB + k0 + 32);
      __builtin_prefetch(gA0 + k0 + 64, 0, 3);   // k+2 tile into near cache
    }
    const bf16* pa = sA[0];
    const bf16* pb = sB[0];
#endif

    v16bf af[2], bfr[2];
#pragma unroll
    for (int t = 0; t < 2; ++t) {
      const bf16* pA = &pa[(wm * 32 + t * 16 + ar) * 40 + ak];
      af[t] = cat8(*(const v8bf*)pA, *(const v8bf*)(pA + 16));
      const bf16* pB = &pb[(wn * 32 + t * 16 + ar) * 40 + bk];
      bfr[t] = cat8(*(const v8bf*)pB, *(const v8bf*)(pB + 16));
    }
#pragma unroll
    for (int i = 0; i < 2; ++i)
#pragma unroll
      for (int j = 0; j < 2; ++j)
        acc[i][j] = __builtin_amdgcn_wmma_f32_16x16x32_bf16(
            false, af[i], false, bfr[j], (short)0, acc[i][j], false, false);
    __syncthreads();
#if ASYNC_LDS
    buf ^= 1;
#endif
  }

  // ---- epilogue: C/D layout = VGPR r -> row r (+8 for lanes 16..31), col=lane&15
  const int row0 = bm + wm * 32;
  const int col0 = bn + wn * 32;
  const int mh   = (lane >> 4) << 3;
#pragma unroll
  for (int i = 0; i < 2; ++i) {
#pragma unroll
    for (int j = 0; j < 2; ++j) {
      const int col = col0 + j * 16 + ar;
      const float bv = BIAS ? bias[col] : 0.f;
#pragma unroll
      for (int r = 0; r < 8; ++r) {
        const int row = row0 + i * 16 + mh + r;
        float v = acc[i][j][r] * alpha + bv;
        if (RELU) v = fmaxf(v, 0.f);
        const long off = cz + (long)row * ldc + col;
        if (OUTF32)  Cf[off] = v;
        if (OUTBF16) Cb[off] = (bf16)v;
      }
    }
  }
}

// ---------------------------------------------------------------------------
extern "C" void kernel_launch(void* const* d_in, const int* in_sizes, int n_in,
                              void* d_out, int out_size, void* d_ws,
                              size_t ws_size, hipStream_t stream) {
  const float* x1  = (const float*)d_in[0];
  const float* x2  = (const float*)d_in[1];
  const float* Wq  = (const float*)d_in[2];
  const float* bq  = (const float*)d_in[3];
  const float* Wk  = (const float*)d_in[4];
  const float* bk  = (const float*)d_in[5];
  const float* Wv  = (const float*)d_in[6];
  const float* bv  = (const float*)d_in[7];
  const float* Wo  = (const float*)d_in[8];
  const float* bo  = (const float*)d_in[9];
  const float* W1  = (const float*)d_in[10];
  const float* b1  = (const float*)d_in[11];
  const float* W2  = (const float*)d_in[12];
  const float* b2  = (const float*)d_in[13];
  const float* g1  = (const float*)d_in[14];
  const float* be1 = (const float*)d_in[15];
  const float* g2  = (const float*)d_in[16];
  const float* be2 = (const float*)d_in[17];
  float* out = (float*)d_out;

  // ---- deterministic workspace carve (~160 MB) ----
  char* p = (char*)d_ws;
  auto carve = [&](size_t bytes) -> void* {
    void* r = (void*)p;
    p += (bytes + 255) & ~(size_t)255;
    return r;
  };
  const long NE    = (long)BSQ * EDIM;   // 2M elems
  const long PLANE = (long)BSQ * BSQ;    // 4M elems per head

  bf16* x1b  = (bf16*)carve(NE * 2);
  bf16* x2b  = (bf16*)carve(NE * 2);
  bf16* wqb  = (bf16*)carve((long)EDIM * EDIM * 2);
  bf16* wkb  = (bf16*)carve((long)EDIM * EDIM * 2);
  bf16* wvb  = (bf16*)carve((long)EDIM * EDIM * 2);
  bf16* wob  = (bf16*)carve((long)EDIM * EDIM * 2);
  bf16* w1b  = (bf16*)carve((long)FF * EDIM * 2);
  bf16* w2b  = (bf16*)carve((long)EDIM * FF * 2);
  bf16* qb   = (bf16*)carve(NE * 2);
  bf16* kb_  = (bf16*)carve(NE * 2);
  bf16* vb_  = (bf16*)carve(NE * 2);
  bf16* vtb  = (bf16*)carve(NE * 2);                 // v^T : [h*128+d][m]
  bf16* sc   = (bf16*)carve((long)NHEAD * PLANE * 2);// scores -> probs (in place)
  bf16* attb = (bf16*)carve(NE * 2);
  float* atto = (float*)carve(NE * 4);
  float* hf   = (float*)carve(NE * 4);
  bf16*  hb   = (bf16*)carve(NE * 2);
  bf16*  f1b  = (bf16*)carve((long)BSQ * FF * 2);
  float* f2   = (float*)carve(NE * 4);

  const dim3 T(256);
  auto cvt = [&](const float* s, bf16* d, long n) {
    cvt_f32_bf16<<<dim3((unsigned)((n + 255) / 256)), T, 0, stream>>>(s, d, n);
  };
  cvt(x1, x1b, NE);
  cvt(x2, x2b, NE);
  cvt(Wq, wqb, (long)EDIM * EDIM);
  cvt(Wk, wkb, (long)EDIM * EDIM);
  cvt(Wv, wvb, (long)EDIM * EDIM);
  cvt(Wo, wob, (long)EDIM * EDIM);
  cvt(W1, w1b, (long)FF * EDIM);
  cvt(W2, w2b, (long)EDIM * FF);

  // ---- Q/K/V projections: [2048,1024] = x @ W^T + b  (bf16 out) ----
  gemm_wmma<true, false, false, true><<<dim3(EDIM / 64, BSQ / 128, 1), T, 0, stream>>>(
      x1b, wqb, bq, nullptr, qb, EDIM, EDIM, EDIM, EDIM, 0, 0, 0, 1.f);
  gemm_wmma<true, false, false, true><<<dim3(EDIM / 64, BSQ / 128, 1), T, 0, stream>>>(
      x2b, wkb, bk, nullptr, kb_, EDIM, EDIM, EDIM, EDIM, 0, 0, 0, 1.f);
  gemm_wmma<true, false, false, true><<<dim3(EDIM / 64, BSQ / 128, 1), T, 0, stream>>>(
      x2b, wvb, bv, nullptr, vb_, EDIM, EDIM, EDIM, EDIM, 0, 0, 0, 1.f);

  // ---- v^T so probs @ v also fits the A @ W^T fragment layout ----
  transpose_bf16<<<dim3(EDIM / 32, BSQ / 32), T, 0, stream>>>(vb_, vtb, BSQ, EDIM);

  // ---- scores[h] = scale * q_h @ k_h^T  (batched over heads via z) ----
  const float scale = 1.0f / sqrtf((float)HDIM);
  gemm_wmma<false, false, false, true><<<dim3(BSQ / 64, BSQ / 128, NHEAD), T, 0, stream>>>(
      qb, kb_, nullptr, nullptr, sc,
      HDIM, EDIM, EDIM, BSQ, (long)HDIM, (long)HDIM, PLANE, scale);

  // ---- softmax across the 8 heads (pointwise) ----
  softmax_heads<<<dim3((unsigned)(PLANE / 256)), T, 0, stream>>>(sc, PLANE);

  // ---- attn[n, h*128+d] = probs_h @ v_h  (batched over heads) ----
  gemm_wmma<false, false, false, true><<<dim3(HDIM / 64, BSQ / 128, NHEAD), T, 0, stream>>>(
      sc, vtb, nullptr, nullptr, attb,
      BSQ, BSQ, BSQ, EDIM, PLANE, (long)HDIM * BSQ, (long)HDIM, 1.f);

  // ---- output projection (f32 out) ----
  gemm_wmma<true, false, true, false><<<dim3(EDIM / 64, BSQ / 128, 1), T, 0, stream>>>(
      attb, wob, bo, atto, nullptr, EDIM, EDIM, EDIM, EDIM, 0, 0, 0, 1.f);

  // ---- h = LN(x1 + attn) ----
  add_layernorm<true><<<dim3(BSQ), T, 0, stream>>>(x1, atto, g1, be1, hf, hb);

  // ---- FFN: relu(h @ W1^T + b1) -> bf16 ; @ W2^T + b2 -> f32 ----
  gemm_wmma<true, true, false, true><<<dim3(FF / 64, BSQ / 128, 1), T, 0, stream>>>(
      hb, w1b, b1, nullptr, f1b, EDIM, EDIM, EDIM, FF, 0, 0, 0, 1.f);
  gemm_wmma<true, false, true, false><<<dim3(EDIM / 64, BSQ / 128, 1), T, 0, stream>>>(
      f1b, w2b, b2, f2, nullptr, FF, FF, FF, EDIM, 0, 0, 0, 1.f);

  // ---- out = LN(h + ffn) ----
  add_layernorm<false><<<dim3(BSQ), T, 0, stream>>>(hf, f2, g2, be2, out, nullptr);
}
